// TE_39453569581548
// MI455X (gfx1250) — matrix operand order repaired and checked by reading, not compiled, not run on hardware
//
#include <hip/hip_runtime.h>

typedef __attribute__((ext_vector_type(16))) _Float16 v16h;
typedef __attribute__((ext_vector_type(4)))  _Float16 h4;
typedef __attribute__((ext_vector_type(8)))  float    v8f;
typedef __attribute__((ext_vector_type(4)))  float    f4;
typedef __attribute__((ext_vector_type(4)))  int      si4;
typedef __attribute__((ext_vector_type(8)))  int      si8;
typedef __attribute__((ext_vector_type(4)))  unsigned int u32x4;

#define LL     64
#define MPL    2048
#define DD     128
#define FOURD  512
#define PP     4
#define SS     2
#define BR     32        // rows per block
#define SRX    136       // padded row stride in f16 elements (272B = 16B-aligned)

// LDS partition (bytes) -- all offsets multiples of 16
#define W_BYTES   (FOURD * SRX * 2)          // 139264
#define XH_BYTES  (BR * SRX * 2)             // 8704
#define OFF_WT    0
#define OFF_UT    (W_BYTES)
#define OFF_XB    (2 * W_BYTES)
#define OFF_HM    (2 * W_BYTES + XH_BYTES)
#define OFF_HS    (2 * W_BYTES + 2 * XH_BYTES)
#define LDS_TOTAL (2 * W_BYTES + 3 * XH_BYTES)   // 304640 < 320K

union FragU { u32x4 q[2]; v16h v; };
static_assert(sizeof(v16h) == 32, "frag size");

// A-matrix 16x32 f16 fragment from row-major LDS tile (stride SRX halfwords).
// ISA layout: lane (r=lane&15, h=lane>>4): VGPR0-3 = K = k0+8h+0..7 (16B),
// VGPR4-7 = K = k0+8h+16..23 (16B). Two contiguous b128 loads.
__device__ __forceinline__ v16h load_a_frag(const _Float16* base, int lane, int k0) {
  const int half = lane >> 4, r = lane & 15;
  const _Float16* p = base + r * SRX + 8 * half + k0;
  FragU f;
  f.q[0] = *(const u32x4*)(p);
  f.q[1] = *(const u32x4*)(p + 16);
  return f.v;
}

// B-matrix 32x16 f16 fragment from transposed weights Wt[N][K] (stride SRX halfwords).
// ISA layout: lane (n=lane&15, h=lane>>4) holds K = k0+8h+0..7 and k0+8h+16..23 (two 16B runs).
__device__ __forceinline__ v16h load_b_frag(const _Float16* wt, int lane, int coltile, int k0) {
  const int half = lane >> 4, n = (lane & 15) + coltile * 16;
  const _Float16* p = wt + n * SRX + k0 + 8 * half;
  FragU f;
  f.q[0] = *(const u32x4*)(p);
  f.q[1] = *(const u32x4*)(p + 16);
  return f.v;
}

#if __has_builtin(__builtin_amdgcn_rcpf)
#define FRCP(x) __builtin_amdgcn_rcpf(x)
#else
#define FRCP(x) (1.f / (x))
#endif

__device__ __forceinline__ float sigm(float x) {
  return FRCP(1.f + __expf(-x));                 // v_rcp + v_exp
}

__device__ __forceinline__ float fast_tanh(float x) {
#if __has_builtin(__builtin_amdgcn_tanhf)
  return __builtin_amdgcn_tanhf(x);              // hardware V_TANH_F32
#else
  return 2.f * FRCP(1.f + __expf(-2.f * x)) - 1.f;
#endif
}

// TDM: stage one f16 weight matrix [512 rows][128 cols] (row stride 256B in global)
// into LDS at byte offset lds_off with 16B of padding appended per 256B row
// (padded LDS row stride = 272B = SRX halfwords).
__device__ __forceinline__ void tdm_load_weight(const _Float16* g, uint32_t lds_off) {
  uint64_t ga = (uint64_t)(uintptr_t)g;
  // D# group 0: count=1 | lds_addr | global_addr[56:0] | type=2
  si4 g0 = { 1,
             (int)lds_off,
             (int)(uint32_t)(ga & 0xFFFFFFFFu),
             (int)(((uint32_t)(ga >> 32) & 0x01FFFFFFu) | (2u << 30)) };
  // D# group 1: data_size=2B, pad_enable, pad_interval=64 DWORDs (code 5),
  // pad_amount=4 DWORDs (code 3); tensor_dim0=128, tensor_dim1=512,
  // tile 128x512, tensor_dim0_stride=128.
  si8 g1 = { (1 << 16) | (1 << 20) | (5 << 22) | (3 << 25),
             (DD & 0xFFFF) << 16,
             (FOURD & 0xFFFF) << 16,
             (DD & 0xFFFF) << 16,
             FOURD,
             DD,
             0,
             0 };
  asm volatile("tensor_load_to_lds %0, %1" :: "s"(g0), "s"(g1) : "memory");
}

// Prep: convert fp32 W[K=128][N=512] to f16 transposed Wt[N=512][K=128] (unpadded in global)
__global__ void convert_transpose_kernel(const float* __restrict__ W, _Float16* __restrict__ Wt) {
  int i = blockIdx.x * blockDim.x + threadIdx.x;
  if (i < FOURD * DD) {
    int n = i >> 7, k = i & 127;
    Wt[i] = (_Float16)W[k * FOURD + n];
  }
}

// Layer 0: emb[0:M] = basic_table[basic_names]
__global__ void layer0_kernel(const int* __restrict__ names, const float* __restrict__ table,
                              float* __restrict__ emb) {
  int i = blockIdx.x * blockDim.x + threadIdx.x;
  if (i < MPL * DD) {
    int r = i >> 7, c = i & 127;
    emb[i] = table[names[r] * DD + c];
  }
}

__global__ __launch_bounds__(256) void layer_kernel(
    float* __restrict__ emb,
    const int* __restrict__ prop_ids, const int* __restrict__ super_ids,
    const _Float16* __restrict__ Wt_m, const _Float16* __restrict__ Ut_m,
    const _Float16* __restrict__ Wt_s, const _Float16* __restrict__ Ut_s,
    const float* __restrict__ b_m, const float* __restrict__ b_s,
    const float* __restrict__ empty_members, int layer) {
  extern __shared__ char smem[];
  _Float16* Wt = (_Float16*)(smem + OFF_WT);   // [512][SRX] current W (transposed)
  _Float16* Ut = (_Float16*)(smem + OFF_UT);   // [512][SRX] current U (transposed)
  _Float16* xb = (_Float16*)(smem + OFF_XB);   // [32][SRX] gathered x rows
  _Float16* hm = (_Float16*)(smem + OFF_HM);   // [32][SRX] members hidden state
  _Float16* hs = (_Float16*)(smem + OFF_HS);   // [32][SRX] supertype hidden state

  const int tid  = threadIdx.x;
  const int lane = tid & 31;
  const int wid  = tid >> 5;               // wave id == column group (16 cols of D)
  const int row0 = blockIdx.x * BR;        // first row of this block within the layer
  const int nn   = lane & 15;
  const int half = lane >> 4;

  // zero hidden-state buffers (padding included), 4B at a time
  {
    uint32_t* hm32 = (uint32_t*)hm;
    uint32_t* hs32 = (uint32_t*)hs;
    for (int i = tid; i < BR * SRX / 2; i += 256) {
      hm32[i] = 0u;
      hs32[i] = 0u;
    }
  }

  auto stage_weights = [&](const _Float16* Wg, const _Float16* Ug) {
    if (tid == 0) {
      tdm_load_weight(Wg, OFF_WT);
      tdm_load_weight(Ug, OFF_UT);
    }
    if (wid == 0) __builtin_amdgcn_s_wait_tensorcnt(0);   // issuing wave drains TENSORcnt
    // visibility to all waves via the next __syncthreads()
  };

  // vectorized gather: 4 cols per thread-iteration (float4 load, 4xf16 store)
  auto fill_x_gather = [&](const int* ids, int stride, int t) {
    for (int i = tid; i < BR * (DD / 4); i += 256) {
      int r = i >> 5, c4 = (i & 31) * 4;
      int g = ids[((size_t)layer * MPL + row0 + r) * stride + t];  // index < layer*M
      f4 v = *(const f4*)&emb[(size_t)g * DD + c4];
      h4 o = { (_Float16)v.x, (_Float16)v.y, (_Float16)v.z, (_Float16)v.w };
      *(h4*)&xb[r * SRX + c4] = o;
    }
  };

  auto fill_x_bcast = [&](const float* v) {
    for (int i = tid; i < BR * (DD / 4); i += 256) {
      int r = i >> 5, c4 = (i & 31) * 4;
      f4 w = *(const f4*)&v[c4];
      h4 o = { (_Float16)w.x, (_Float16)w.y, (_Float16)w.z, (_Float16)w.w };
      *(h4*)&xb[r * SRX + c4] = o;
    }
  };

  auto prefetch_rows = [&](const int* ids, int stride, int t) {
    for (int r = tid; r < BR; r += 256) {
      int g = ids[((size_t)layer * MPL + row0 + r) * stride + t];
      const float* p = &emb[(size_t)g * DD];
      __builtin_prefetch(p, 0, 0);
      __builtin_prefetch(p + 64, 0, 0);
    }
  };

  // One LSTM timestep for this wave's 16-column group over BOTH row tiles.
  // B fragments loaded once per (gate,kf) and reused for the two row tiles;
  // next B pair preloaded while current four WMMAs issue (1-stage pipeline).
  auto lstm_step = [&](const _Float16* asrc, const _Float16* hb, const float* bv,
                       v8f& c0, v8f& c1, v8f& h0, v8f& h1) {
    v16h ax0[4], ah0[4], ax1[4], ah1[4];
#pragma unroll
    for (int kf = 0; kf < 4; ++kf) {
      ax0[kf] = load_a_frag(asrc,            lane, kf * 32);
      ax1[kf] = load_a_frag(asrc + 16 * SRX, lane, kf * 32);
      ah0[kf] = load_a_frag(hb,              lane, kf * 32);
      ah1[kf] = load_a_frag(hb + 16 * SRX,   lane, kf * 32);
    }
    v8f z0[4], z1[4];
    v16h bw = load_b_frag(Wt, lane, wid, 0);          // gate 0, kf 0
    v16h bu = load_b_frag(Ut, lane, wid, 0);
#pragma unroll
    for (int gate = 0; gate < 4; ++gate) {
      const float b = bv[gate];
      v8f a0 = {b, b, b, b, b, b, b, b};
      v8f a1 = a0;
#pragma unroll
      for (int kf = 0; kf < 4; ++kf) {
        v16h cw = bw, cu = bu;
        if (!(gate == 3 && kf == 3)) {                // preload next B pair
          const int ng = (kf == 3) ? gate + 1 : gate;
          const int nk = (kf == 3) ? 0 : kf + 1;
          bw = load_b_frag(Wt, lane, ng * 8 + wid, nk * 32);
          bu = load_b_frag(Ut, lane, ng * 8 + wid, nk * 32);
        }
        a0 = __builtin_amdgcn_wmma_f32_16x16x32_f16(false, ax0[kf], false, cw,
                                                    (short)0, a0, false, false);
        a1 = __builtin_amdgcn_wmma_f32_16x16x32_f16(false, ax1[kf], false, cw,
                                                    (short)0, a1, false, false);
        a0 = __builtin_amdgcn_wmma_f32_16x16x32_f16(false, ah0[kf], false, cu,
                                                    (short)0, a0, false, false);
        a1 = __builtin_amdgcn_wmma_f32_16x16x32_f16(false, ah1[kf], false, cu,
                                                    (short)0, a1, false, false);
      }
      z0[gate] = a0;
      z1[gate] = a1;
    }
#pragma unroll
    for (int e = 0; e < 8; ++e) {
      float si = sigm(z0[0][e]), sf = sigm(z0[1][e]);
      float tg = fast_tanh(z0[2][e]), so = sigm(z0[3][e]);
      float cc = sf * c0[e] + si * tg;
      c0[e] = cc;
      h0[e] = so * fast_tanh(cc);
    }
#pragma unroll
    for (int e = 0; e < 8; ++e) {
      float si = sigm(z1[0][e]), sf = sigm(z1[1][e]);
      float tg = fast_tanh(z1[2][e]), so = sigm(z1[3][e]);
      float cc = sf * c1[e] + si * tg;
      c1[e] = cc;
      h1[e] = so * fast_tanh(cc);
    }
  };

  auto write_h = [&](_Float16* hb, const v8f& h, int rowtile) {
    _Float16* hbase = hb + rowtile * 16 * SRX;
#pragma unroll
    for (int e = 0; e < 8; ++e) {
      int m = e + 8 * half;
      hbase[m * SRX + wid * 16 + nn] = (_Float16)h[e];
    }
  };

  auto store_global = [&](const v8f& h, int rowtile) {
#pragma unroll
    for (int e = 0; e < 8; ++e) {
      int m = rowtile * 16 + e + 8 * half;
      emb[((size_t)layer * MPL + row0 + m) * DD + wid * 16 + nn] = h[e];
    }
  };

  // ---------------- members LSTM: 5 steps ----------------
  v8f c0 = {}, c1 = {}, h0, h1;
  float bvm[4], bvs[4];
#pragma unroll
  for (int g = 0; g < 4; ++g) {
    bvm[g] = b_m[(g * 8 + wid) * 16 + nn];
    bvs[g] = b_s[(g * 8 + wid) * 16 + nn];
  }
  stage_weights(Wt_m, Ut_m);
  prefetch_rows(prop_ids, PP, 0);            // rows needed at t=1
  for (int t = 0; t < PP + 1; ++t) {
    if (t == 0) fill_x_bcast(empty_members);
    else        fill_x_gather(prop_ids, PP, t - 1);
    if (t < PP) prefetch_rows(prop_ids, PP, t);   // prefetch next step's gather
    __syncthreads();                         // x, weights, previous-h all visible
    lstm_step(xb, hm, bvm, c0, c1, h0, h1);
    __syncthreads();                         // all reads of old h complete
    write_h(hm, h0, 0);
    write_h(hm, h1, 1);
    __syncthreads();                         // new h visible to all waves
  }

  // ---------------- supertypes LSTM: 3 steps ----------------
  c0 = (v8f){};
  c1 = (v8f){};
  stage_weights(Wt_s, Ut_s);
  prefetch_rows(super_ids, SS, 0);
  for (int t = 0; t < SS + 1; ++t) {
    const _Float16* asrc;
    if (t < SS) { fill_x_gather(super_ids, SS, t); asrc = xb; }
    else        { asrc = hm; }               // last input is h_m (already in LDS)
    if (t + 1 < SS) prefetch_rows(super_ids, SS, t + 1);
    __syncthreads();
    lstm_step(asrc, hs, bvs, c0, c1, h0, h1);
    __syncthreads();
    write_h(hs, h0, 0);
    write_h(hs, h1, 1);
    if (t == SS) {                           // h_s -> emb[layer*M : (layer+1)*M]
      store_global(h0, 0);
      store_global(h1, 1);
    }
    __syncthreads();
  }
}

extern "C" void kernel_launch(void* const* d_in, const int* in_sizes, int n_in,
                              void* d_out, int out_size, void* d_ws, size_t ws_size,
                              hipStream_t stream) {
  const int*   basic_names   = (const int*)d_in[0];
  const int*   prop_ids      = (const int*)d_in[1];
  const int*   super_ids     = (const int*)d_in[2];
  const float* basic_table   = (const float*)d_in[3];
  const float* W_m           = (const float*)d_in[4];
  const float* U_m           = (const float*)d_in[5];
  const float* b_m           = (const float*)d_in[6];
  const float* W_s           = (const float*)d_in[7];
  const float* U_s           = (const float*)d_in[8];
  const float* b_s           = (const float*)d_in[9];
  const float* empty_members = (const float*)d_in[10];
  float* emb = (float*)d_out;

  // workspace: 4 weight matrices, f16, transposed [512][128]
  _Float16* Wt_m = (_Float16*)d_ws;
  _Float16* Ut_m = Wt_m + FOURD * DD;
  _Float16* Wt_s = Ut_m + FOURD * DD;
  _Float16* Ut_s = Wt_s + FOURD * DD;

  const int tpb = 256;
  const int wblk = (FOURD * DD + tpb - 1) / tpb;
  convert_transpose_kernel<<<wblk, tpb, 0, stream>>>(W_m, Wt_m);
  convert_transpose_kernel<<<wblk, tpb, 0, stream>>>(U_m, Ut_m);
  convert_transpose_kernel<<<wblk, tpb, 0, stream>>>(W_s, Wt_s);
  convert_transpose_kernel<<<wblk, tpb, 0, stream>>>(U_s, Ut_s);

  layer0_kernel<<<(MPL * DD + tpb - 1) / tpb, tpb, 0, stream>>>(basic_names, basic_table, emb);

  for (int l = 1; l < LL; ++l) {
    layer_kernel<<<MPL / BR, 256, LDS_TOTAL, stream>>>(
        emb, prop_ids, super_ids, Wt_m, Ut_m, Wt_s, Ut_s, b_m, b_s, empty_members, l);
  }
}
